// CascadedBranch_dynamic_7524782703179
// MI455X (gfx1250) — matrix operand order repaired and checked by reading, not compiled, not run on hardware
//
#include <hip/hip_runtime.h>

typedef __attribute__((ext_vector_type(16))) _Float16 v16h;
typedef __attribute__((ext_vector_type(8)))  _Float16 v8h;
typedef __attribute__((ext_vector_type(8)))  float    v8f;
typedef unsigned int u32;
typedef __attribute__((ext_vector_type(4))) u32 u32x4;
typedef __attribute__((ext_vector_type(8))) int i32x8;
typedef __attribute__((ext_vector_type(4))) int i32x4;

#define BB 8
#define TT 1024
#define AA 768
#define DD 512
#define VV 49408
#define SS 48
#define MM (BB*SS)           // 384
#define MTILES (MM/16)       // 24
#define NVS 8                // V-stream splits
#define EPSF 1e-8f
#define NEG_BIG -3.0e38f

#if defined(__has_builtin)
#if __has_builtin(__builtin_amdgcn_tensor_load_to_lds) && __has_builtin(__builtin_amdgcn_s_wait_tensorcnt)
#define USE_TDM 1
#endif
#endif

// Build WMMA A-operand chunk (16x32 f16) from a row-major row of halves.
// Layout: lanes 0-15 hold K {0..15}; lanes 16-31 hold K {16..31} (contiguous, matches B).
__device__ __forceinline__ v16h load_a16(const _Float16* rowp, int kbase, int h) {
  v8h lo = *(const v8h*)(rowp + kbase + h * 16);
  v8h hi = *(const v8h*)(rowp + kbase + h * 16 + 8);
  return __builtin_shufflevector(lo, hi, 0,1,2,3,4,5,6,7,8,9,10,11,12,13,14,15);
}

// Copy a 16x512 f16 tile (16 KB, contiguous) from global to LDS.
// Preferred path: Tensor Data Mover (1-D descriptor, one instruction).
__device__ __forceinline__ void tile_load_16x512(_Float16* dstl, const _Float16* srcg,
                                                 int tid, int nthr) {
#if USE_TDM
  if (tid == 0) {
    u32 lds = (u32)(size_t)(__attribute__((address_space(3))) _Float16*)dstl;
    unsigned long long ga = (unsigned long long)(size_t)srcg;
    u32x4 g0 = { 1u,                                    // count=1, user mode
                 lds,                                   // lds_addr
                 (u32)ga,                               // global_addr[31:0]
                 ((u32)(ga >> 32) & 0x01FFFFFFu) | (2u << 30) }; // ga[56:32] | type=2
    i32x8 g1 = { (int)(1u << 16),                       // data_size=1 (2 bytes), mask=0
                 (int)(0x2000u << 16),                  // tensor_dim0 = 8192 (bits 63:48)
                 (int)(1u << 16),                       // tensor_dim1 = 1
                 (int)(0x2000u << 16),                  // tile_dim0 = 8192
                 0,                                     // tile_dim1/2 = 0 (unused)
                 8192,                                  // tensor_dim0_stride
                 0, 0 };
    i32x4 z4 = { 0, 0, 0, 0 };
    i32x8 z8 = { 0, 0, 0, 0, 0, 0, 0, 0 };
    __builtin_amdgcn_tensor_load_to_lds(g0, g1, z4, z4, z8, 0);
  }
  __builtin_amdgcn_s_wait_tensorcnt(0);
#else
  const u32* s = (const u32*)srcg;
  u32* d = (u32*)dstl;
  for (int i = tid; i < 16 * DD / 2; i += nthr) d[i] = s[i];
#endif
}

// Score tile: C = kwA(16x512) . E16[token]^T with 4-deep B prefetch.
__device__ __forceinline__ v8f score_tile(const v16h* a, const _Float16* bp) {
  v8f c = {};
  v16h bbuf[4];
  #pragma unroll
  for (int p = 0; p < 4; ++p) bbuf[p] = *(const v16h*)(bp + p * 32);
  #pragma unroll
  for (int kc = 0; kc < 16; ++kc) {
    v16h cur = bbuf[kc & 3];
    if (kc < 12) bbuf[kc & 3] = *(const v16h*)(bp + (kc + 4) * 32);
    c = __builtin_amdgcn_wmma_f32_16x16x32_f16(false, a[kc], false, cur,
                                               (short)0, c, false, false);
  }
  return c;
}

// ---------------- K1: CIF downsample + projection + L2-normalize -> kw16 ----
__global__ __launch_bounds__(256)
void k_cif_proj(const float* __restrict__ audio, const float* __restrict__ W,
                const float* __restrict__ bias, const int* __restrict__ fp,
                _Float16* __restrict__ kw16) {
  __shared__ float feat[AA];
  __shared__ float red[8];
  int bs = blockIdx.x;
  int b = bs / SS, s = bs % SS;
  int end = fp[b * SS + s];
  int start = (s == 0) ? 0 : fp[b * SS + s - 1];
  if (end > TT) end = TT; if (end < 0) end = 0;
  if (start > TT) start = TT; if (start < 0) start = 0;
  int len = end - start; if (len < 0) len = 0;
  float inv = (len > 0) ? (1.0f / (float)len) : 0.0f;
  int tid = threadIdx.x;

  for (int a = tid; a < AA; a += 256) {
    float acc = 0.f;
    for (int t = start; t < end; ++t)
      acc += audio[((size_t)b * TT + t) * AA + a];
    feat[a] = acc * inv;
  }
  __syncthreads();

  float o0 = bias[tid], o1 = bias[tid + 256];
  for (int a = 0; a < AA; ++a) {
    float f = feat[a];
    o0 = fmaf(f, W[(size_t)a * DD + tid], o0);
    o1 = fmaf(f, W[(size_t)a * DD + tid + 256], o1);
  }
  float ss = o0 * o0 + o1 * o1;
  #pragma unroll
  for (int d = 16; d >= 1; d >>= 1) ss += __shfl_xor(ss, d, 32);
  if ((tid & 31) == 0) red[tid >> 5] = ss;
  __syncthreads();
  if (tid == 0) {
    float t = 0.f;
    #pragma unroll
    for (int i = 0; i < 8; ++i) t += red[i];
    red[0] = 1.0f / fmaxf(sqrtf(t), EPSF);
  }
  __syncthreads();
  float sc = red[0];
  kw16[(size_t)bs * DD + tid]       = (_Float16)(o0 * sc);
  kw16[(size_t)bs * DD + tid + 256] = (_Float16)(o1 * sc);
}

// ---------------- K2: token_emb -> f16 copy + inverse norms -----------------
__global__ __launch_bounds__(256)
void k_emb_prep(const float* __restrict__ E, _Float16* __restrict__ E16,
                float* __restrict__ inv_en) {
  __shared__ float red[8];
  int v = blockIdx.x, tid = threadIdx.x;
  float e0 = E[(size_t)v * DD + tid];
  float e1 = E[(size_t)v * DD + tid + 256];
  E16[(size_t)v * DD + tid]       = (_Float16)e0;
  E16[(size_t)v * DD + tid + 256] = (_Float16)e1;
  float ss = e0 * e0 + e1 * e1;
  #pragma unroll
  for (int d = 16; d >= 1; d >>= 1) ss += __shfl_xor(ss, d, 32);
  if ((tid & 31) == 0) red[tid >> 5] = ss;
  __syncthreads();
  if (tid == 0) {
    float t = 0.f;
    #pragma unroll
    for (int i = 0; i < 8; ++i) t += red[i];
    inv_en[v] = 1.0f / fmaxf(sqrtf(t), EPSF);
  }
}

// ------- K3: single score pass -> store f16 scores + partial stats ----------
__global__ __launch_bounds__(256)
void k_stats(const _Float16* __restrict__ kw16, const _Float16* __restrict__ E16,
             const float* __restrict__ inv_en, _Float16* __restrict__ P,
             float* __restrict__ pmws, float* __restrict__ plws) {
  __shared__ alignas(32) _Float16 kwl[16 * DD];
  __shared__ float pm[8][16];
  __shared__ float pl[8][16];
  int tid = threadIdx.x, mt = blockIdx.x, gy = blockIdx.y;

  tile_load_16x512(kwl, kw16 + (size_t)mt * 16 * DD, tid, 256);
  __syncthreads();

  int lane = tid & 31, wave = tid >> 5;
  int row = lane & 15, h = lane >> 4;

  v16h a[16];
  #pragma unroll
  for (int kc = 0; kc < 16; ++kc) a[kc] = load_a16(kwl + row * DD, kc * 32, h);

  float m8[8], l8[8];
  #pragma unroll
  for (int r = 0; r < 8; ++r) { m8[r] = NEG_BIG; l8[r] = 0.f; }

  for (int j = gy * 8 + wave; j < VV / 16; j += 8 * NVS) {
    int token = j * 16 + (lane & 15);
    v8f c = score_tile(a, E16 + (size_t)token * DD + h * 16);
    float se = inv_en[token] * 10.0f;   // 1/temperature
    #pragma unroll
    for (int r = 0; r < 8; ++r) {
      _Float16 s16 = (_Float16)(c[r] * se);       // quantize once...
      P[(size_t)(mt * 16 + h * 8 + r) * VV + token] = s16;
      float sv = (float)s16;                      // ...and keep stats consistent with it
      float nm = fmaxf(m8[r], sv);
      l8[r] = l8[r] * __expf(m8[r] - nm) + __expf(sv - nm);
      m8[r] = nm;
    }
  }
  #pragma unroll
  for (int d = 1; d < 16; d <<= 1) {
    #pragma unroll
    for (int r = 0; r < 8; ++r) {
      float om = __shfl_xor(m8[r], d, 32);
      float ol = __shfl_xor(l8[r], d, 32);
      float nm = fmaxf(m8[r], om);
      l8[r] = l8[r] * __expf(m8[r] - nm) + ol * __expf(om - nm);
      m8[r] = nm;
    }
  }
  if ((lane & 15) == 0) {
    #pragma unroll
    for (int r = 0; r < 8; ++r) { pm[wave][h * 8 + r] = m8[r]; pl[wave][h * 8 + r] = l8[r]; }
  }
  __syncthreads();
  if (tid < 16) {
    float m = NEG_BIG, l = 0.f;
    #pragma unroll
    for (int w = 0; w < 8; ++w) {
      float om = pm[w][tid], ol = pl[w][tid];
      float nm = fmaxf(m, om);
      l = l * __expf(m - nm) + ol * __expf(om - nm);
      m = nm;
    }
    pmws[(size_t)gy * MM + mt * 16 + tid] = m;
    plws[(size_t)gy * MM + mt * 16 + tid] = l;
  }
}

// ---------------- K3b: combine partial stats across the 8 V-splits ---------
__global__ __launch_bounds__(128)
void k_stats_combine(const float* __restrict__ pmws, const float* __restrict__ plws,
                     float* __restrict__ mrow, float* __restrict__ lrow) {
  int r = blockIdx.x * 128 + threadIdx.x;
  if (r >= MM) return;
  float m = NEG_BIG, l = 0.f;
  #pragma unroll
  for (int g = 0; g < NVS; ++g) {
    float om = pmws[(size_t)g * MM + r];
    float ol = plws[(size_t)g * MM + r];
    float nm = fmaxf(m, om);
    l = l * __expf(m - nm) + ol * __expf(om - nm);
    m = nm;
  }
  mrow[r] = m;
  lrow[r] = l;
}

// ---------------- K3c: in-place P <- exp(P - m[row]) ------------------------
__global__ __launch_bounds__(256)
void k_pexp(_Float16* __restrict__ P, const float* __restrict__ mrow) {
  int idx = blockIdx.x * 256 + threadIdx.x;     // u32 index into P
  int rowr = (int)(((long long)idx * 2) / VV);
  float m = mrow[rowr];
  u32* p32 = (u32*)P;
  u32 pr = p32[idx];
  float a = (float)((_Float16*)&pr)[0];
  float b = (float)((_Float16*)&pr)[1];
  _Float16 ea = (_Float16)__expf(a - m);
  _Float16 eb = (_Float16)__expf(b - m);
  u32 outv = (u32)*(unsigned short*)&ea | ((u32)*(unsigned short*)&eb << 16);
  p32[idx] = outv;
}

// ---------------- K4: pure GEMM pout += P(16xV) @ E(Vx512) via WMMA ---------
__global__ __launch_bounds__(128)
void k_readout(const _Float16* __restrict__ P, const _Float16* __restrict__ E16,
               float* __restrict__ pout) {
  __shared__ alignas(32) _Float16 etl[128 * 144];   // E tile transposed [col][token]
  int tid = threadIdx.x, mt = blockIdx.x;
  int cslice = blockIdx.y * 128;
  int gz = blockIdx.z;

  int lane = tid & 31, wave = tid >> 5;
  int h = lane >> 4;
  const _Float16* prow = P + (size_t)(mt * 16 + (lane & 15)) * VV;

  v8f acc0 = {}, acc1 = {};

  for (int i = gz; i < VV / 128; i += NVS) {
    int v0 = i * 128;
    // ---- stage E[v0..v0+127][cslice..cslice+127] transposed into LDS ----
    for (int q = 0; q < 64; ++q) {
      int lin = q * 128 + tid;
      int tok = lin >> 6;
      int cp  = lin & 63;
      u32 pr = *(const u32*)(E16 + (size_t)(v0 + tok) * DD + cslice + cp * 2);
      _Float16 lo = ((const _Float16*)&pr)[0];
      _Float16 hi = ((const _Float16*)&pr)[1];
      etl[(size_t)(cp * 2) * 144 + tok]     = lo;
      etl[(size_t)(cp * 2 + 1) * 144 + tok] = hi;
    }
    __syncthreads();
    // ---- acc += P(16x128) @ E(128 x my 32 cols) ----
    #pragma unroll
    for (int kc2 = 0; kc2 < 4; ++kc2) {
      v16h ap = load_a16(prow, v0 + kc2 * 32, h);
      int c0 = wave * 32 + (lane & 15);
      v16h b0 = *(const v16h*)(etl + (size_t)c0 * 144 + kc2 * 32 + h * 16);
      v16h b1 = *(const v16h*)(etl + (size_t)(c0 + 16) * 144 + kc2 * 32 + h * 16);
      acc0 = __builtin_amdgcn_wmma_f32_16x16x32_f16(false, ap, false, b0,
                                                    (short)0, acc0, false, false);
      acc1 = __builtin_amdgcn_wmma_f32_16x16x32_f16(false, ap, false, b1,
                                                    (short)0, acc1, false, false);
    }
    __syncthreads();
  }

  float* pp = pout + (size_t)gz * MM * DD;
  #pragma unroll
  for (int r = 0; r < 8; ++r) {
    int rg = mt * 16 + h * 8 + r;
    int cg = cslice + wave * 32 + (lane & 15);
    pp[(size_t)rg * DD + cg]      = acc0[r];
    pp[(size_t)rg * DD + cg + 16] = acc1[r];
  }
}

// ---------------- K4b: sum 8 partials, scale by 1/l ------------------------
__global__ __launch_bounds__(256)
void k_combine(const float* __restrict__ pout, const float* __restrict__ lrow,
               float* __restrict__ out) {
  int idx = blockIdx.x * 256 + threadIdx.x;     // < MM*DD
  int rg = idx >> 9;
  float s = 0.f;
  #pragma unroll
  for (int g = 0; g < NVS; ++g) s += pout[(size_t)g * MM * DD + idx];
  out[idx] = s / lrow[rg];
}

// ---------------------------------------------------------------------------
extern "C" void kernel_launch(void* const* d_in, const int* in_sizes, int n_in,
                              void* d_out, int out_size, void* d_ws, size_t ws_size,
                              hipStream_t stream) {
  (void)in_sizes; (void)n_in; (void)out_size; (void)ws_size;
  const float* audio = (const float*)d_in[0];
  const float* W     = (const float*)d_in[1];
  const float* bias  = (const float*)d_in[2];
  const float* emb   = (const float*)d_in[3];
  const int*   fp    = (const int*)d_in[4];
  float* out = (float*)d_out;

  char* ws = (char*)d_ws;
  _Float16* kw16 = (_Float16*)(ws + 0);          // 393216 B
  float*    invn = (float*)(ws + 393216);        // 197632 B
  float*    mr   = (float*)(ws + 590848);        // 1536 B
  float*    lr   = (float*)(ws + 592384);        // 1536 B
  float*    pmws = (float*)(ws + 593920);        // 12288 B
  float*    plws = (float*)(ws + 606208);        // 12288 B
  float*    pout = (float*)(ws + 618496);        // 6291456 B
  _Float16* P    = (_Float16*)(ws + 6909952);    // 37945344 B  (384 x 49408 f16)
  _Float16* E16  = (_Float16*)(ws + 44855296);   // 50593792 B  (total ~95.5 MB)

  k_cif_proj<<<dim3(MM), dim3(256), 0, stream>>>(audio, W, bias, fp, kw16);
  k_emb_prep<<<dim3(VV), dim3(256), 0, stream>>>(emb, E16, invn);
  k_stats<<<dim3(MTILES, NVS), dim3(256), 0, stream>>>(kw16, E16, invn, P, pmws, plws);
  k_stats_combine<<<dim3(3), dim3(128), 0, stream>>>(pmws, plws, mr, lr);
  k_pexp<<<dim3(MM * VV / 2 / 256), dim3(256), 0, stream>>>(P, mr);
  k_readout<<<dim3(MTILES, 4, NVS), dim3(128), 0, stream>>>(P, E16, pout);
  k_combine<<<dim3(MM * DD / 256), dim3(256), 0, stream>>>(pout, lr, out);
}
